// Downsample_30185030157081
// MI455X (gfx1250) — matrix operand order repaired and checked by reading, not compiled, not run on hardware
//
#include <hip/hip_runtime.h>

// ---------------------------------------------------------------------------
// Ternary-weight 3x3 conv, stride 2, pad 1, as implicit GEMM on gfx1250 WMMA.
//   M = Cout = 256, N = B*OH*OW = 32768, K' = 9*Cin = 2304 with
//   k' = (r*3+s)*256 + c   (K permuted so each 32-step is 32 contiguous
//                           channels at fixed (r,s) -> cheap strided gather)
//   Pre-pass: wq[m][k'] = ternarize(w[m][c][r][s]) in f16 (exact for {-1,0,1})
//   Main:     D = wq * im2col(x) + bias, f32 accum via v_wmma_f32_16x16x32_f16
//   A tile   : global->LDS via GLOBAL_LOAD_ASYNC_TO_LDS_B128 (ASYNCcnt).
//   K loop unrolled x8: (r,s) decode/clamp once per octet, B addresses become
//   base + 24-bit immediate offsets, LDS buffer parity folds to constants.
// ---------------------------------------------------------------------------

typedef __attribute__((ext_vector_type(16))) _Float16 v16h;
typedef __attribute__((ext_vector_type(8)))  _Float16 v8h;
typedef __attribute__((ext_vector_type(8)))  float    v8f;
typedef __attribute__((ext_vector_type(4)))  float    v4f;
typedef __attribute__((ext_vector_type(4)))  int      v4i;

constexpr int BATCH = 32;
constexpr int CIN   = 256;
constexpr int HIN   = 64;
constexpr int WIN   = 64;
constexpr int COUT  = 256;
constexpr int OH    = 32;
constexpr int OW    = 32;

constexpr int KDIM   = CIN * 9;          // 2304
constexpr int NTOT   = BATCH * OH * OW;  // 32768
constexpr int MBLK   = 128;
constexpr int NBLK   = 64;
constexpr int KBLK   = 32;
constexpr int KITERS = KDIM / KBLK;      // 72 = 9 (rs) * 8 (channel blocks)
constexpr int LDA    = 40;               // padded halves/row: 80B -> stride-4 banks
constexpr int LDB    = 40;

#define TER_T 0.001f

#if __has_builtin(__builtin_amdgcn_global_load_async_to_lds_b128)
#define HAS_ASYNC_LDS 1
typedef __attribute__((address_space(1))) v4i as1_v4i;   // global 128b chunk
typedef __attribute__((address_space(3))) v4i as3_v4i;   // LDS    128b chunk
#else
#define HAS_ASYNC_LDS 0
#endif

__device__ __forceinline__ void wait_asynccnt0() {
#if __has_builtin(__builtin_amdgcn_s_wait_asynccnt)
    __builtin_amdgcn_s_wait_asynccnt(0);
#else
    asm volatile("s_wait_asynccnt 0x0" ::: "memory");
#endif
}

// ---------------- pre-pass: ternarize + K-permute weights ----------------
// w  : [m][c][rs] fp32  (m*2304 + c*9 + rs)
// wq : [m][rs*256+c] f16
__global__ __launch_bounds__(256)
void quantize_weights(const float* __restrict__ w, _Float16* __restrict__ wq) {
    const int c  = threadIdx.x;   // 0..255
    const int rs = blockIdx.x;    // 0..8
    const int m  = blockIdx.y;    // 0..255
    const float v = w[m * KDIM + c * 9 + rs];
    const float q = v > TER_T ? 1.0f : (v < -TER_T ? -1.0f : 0.0f);
    wq[m * KDIM + (rs << 8) + c] = (_Float16)q;
}

// ---------------- main implicit-GEMM kernel ----------------
__global__ __launch_bounds__(256)
void ternary_conv_wmma(const float* __restrict__ x,
                       const _Float16* __restrict__ wq,
                       const float* __restrict__ bias,
                       float* __restrict__ out) {
    __shared__ __align__(16) _Float16 lA[2][MBLK * LDA];   // 10 KB each
    __shared__ __align__(16) _Float16 lB[2][NBLK * LDB];   // 5 KB each

    const int t    = threadIdx.x;
    const int lane = t & 31;
    const int wave = t >> 5;
    const int nb   = blockIdx.x;          // 0..511  (N blocks of 64)
    const int mb   = blockIdx.y;          // 0..1    (M blocks of 128)

    const int wm = wave & 3;              // wave row quadrant: rows wm*32..+31
    const int wn = wave >> 2;             // wave col half:     cols wn*32..+31

    // Per-lane WMMA fragment geometry (wave32 layouts, CDNA5 ISA 7.12.2)
    const int r16 = lane & 15;
    const int hi  = lane >> 4;            // lane group 0 / 1
    const int akb = hi ? 8  : 0;          // A frag: K = akb..+7 and akb+16..+23
    const int bkb = hi ? 8  : 0;          // B frag: K = bkb..+7 and bkb+16..+23

    // --- A staging map: 2 threads per row, 16 contiguous k' each (f16) ---
    const int arow  = t >> 1;             // 0..127
    const int akseg = (t & 1) << 4;       // 0 or 16
    const _Float16* wrow = wq + (size_t)(mb * MBLK + arow) * KDIM + akseg;

    // --- B staging map: 4 threads per col, 8 channels each ---
    const int bcol  = t & 63;             // 0..63
    const int bkseg = (t >> 6) << 3;      // 0,8,16,24 (channel offset in block)
    const int cg0   = nb * NBLK + bcol;   // global output-position index
    const int bn    = cg0 >> 10;          // image
    const int boh   = (cg0 >> 5) & 31;
    const int bow   = cg0 & 31;
    // x base for this thread's channel sub-block: + cb*32*4096 + ih*64 + iw
    const float* xin = x + ((size_t)(bn * CIN + bkseg) << 12);

#if !HAS_ASYNC_LDS
    v8h   ra0, ra1;
#endif
    float rb[8];
    bool  bvalid = false;

    // A tile: issue global->LDS copies for K-step kt into buffer `buf`.
    auto loadA = [&](int buf, int kt) {
        const int k0 = kt * KBLK;
        __builtin_prefetch(wrow + k0 + KBLK, 0, 0);     // next tile -> caches
#if HAS_ASYNC_LDS
        __builtin_amdgcn_global_load_async_to_lds_b128(
            (as1_v4i*)(wrow + k0),
            (as3_v4i*)&lA[buf][arow * LDA + akseg], 0, 0);
        __builtin_amdgcn_global_load_async_to_lds_b128(
            (as1_v4i*)(wrow + k0 + 8),
            (as3_v4i*)&lA[buf][arow * LDA + akseg + 8], 0, 0);
#else
        const v8h* ap = (const v8h*)(wrow + k0);
        ra0 = ap[0];
        ra1 = ap[1];
        (void)buf;
#endif
    };

    // B tile: (r,s) fixed per octet of kt; 8 channel-strided dword loads.
    // Under x8 unroll, rs is loop-invariant per trip and cb is a constant, so
    // the address folds to one base + immediate offsets.
    auto loadB = [&](int kt) {
        const int rs = kt >> 3;           // kt = rs*8 + cb (constant per trip)
        const int cb = kt & 7;            // constant per unrolled instance
        const int r  = rs / 3;
        const int s  = rs - 3 * r;
        const int ih = 2 * boh + r - 1;
        const int iw = 2 * bow + s - 1;
        bvalid = ((unsigned)ih < (unsigned)HIN) & ((unsigned)iw < (unsigned)WIN);
        const int ihc = ih < 0 ? 0 : (ih > HIN - 1 ? HIN - 1 : ih);
        const int iwc = iw < 0 ? 0 : (iw > WIN - 1 ? WIN - 1 : iw);
        const float* p = xin + ((size_t)(cb << 5) << 12) + (ihc << 6) + iwc;
#pragma unroll
        for (int j = 0; j < 8; ++j) rb[j] = p[(size_t)j << 12];
    };

    // Commit staged B registers (and A on fallback path) into buffer `buf`.
    auto storeB = [&](int buf) {
#if !HAS_ASYNC_LDS
        *(v8h*)&lA[buf][arow * LDA + akseg]     = ra0;
        *(v8h*)&lA[buf][arow * LDA + akseg + 8] = ra1;
#endif
        v8h qb;
#pragma unroll
        for (int j = 0; j < 8; ++j)
            qb[j] = (_Float16)(bvalid ? rb[j] : 0.0f);
        *(v8h*)&lB[buf][bcol * LDB + bkseg] = qb;
    };

    v8f acc[2][2] = {};

    auto compute = [&](int buf) {
        v16h afrag[2], bfrag[2];
#pragma unroll
        for (int i = 0; i < 2; ++i) {
            const _Float16* ap = &lA[buf][(wm * 32 + i * 16 + r16) * LDA + akb];
            v8h alo = *(const v8h*)ap;            // K = akb .. akb+7
            v8h ahi = *(const v8h*)(ap + 16);     // K = akb+16 .. akb+23
            afrag[i] = __builtin_shufflevector(alo, ahi,
                0, 1, 2, 3, 4, 5, 6, 7, 8, 9, 10, 11, 12, 13, 14, 15);
            const _Float16* bp = &lB[buf][(wn * 32 + i * 16 + r16) * LDB + bkb];
            v8h blo = *(const v8h*)bp;
            v8h bhi = *(const v8h*)(bp + 16);
            bfrag[i] = __builtin_shufflevector(blo, bhi,
                0, 1, 2, 3, 4, 5, 6, 7, 8, 9, 10, 11, 12, 13, 14, 15);
        }
#pragma unroll
        for (int ti = 0; ti < 2; ++ti)
#pragma unroll
            for (int tj = 0; tj < 2; ++tj)
                acc[ti][tj] = __builtin_amdgcn_wmma_f32_16x16x32_f16(
                    false, afrag[ti], false, bfrag[tj],
                    (short)0, acc[ti][tj], false, false);
    };

    // -------- software-pipelined main loop (double-buffered LDS) --------
    // Prologue: fill buffer 0 and publish it.
    loadA(0, 0);
    loadB(0);
    storeB(0);
#if HAS_ASYNC_LDS
    wait_asynccnt0();
#endif
    __syncthreads();

#pragma unroll 8
    for (int kt = 0; kt < KITERS; ++kt) {
        const bool more = (kt + 1) < KITERS;
        if (more) {
            // lA[nxt]/lB[nxt] free: compute(kt-1) ds_loads completed before
            // its WMMAs issued, prior to the barrier we just passed.
            loadA((kt + 1) & 1, kt + 1);   // async -> LDS
            loadB(kt + 1);                 // global loads in flight...
        }
        compute(kt & 1);                   // ...overlap with WMMA work
        if (more) {
            storeB((kt + 1) & 1);
#if HAS_ASYNC_LDS
            wait_asynccnt0();              // A tile landed in LDS
#endif
        }
        __syncthreads();                   // publish buf[(kt+1)&1]
    }

    // -------- epilogue: D + bias -> out (NCHW, f32) --------
#pragma unroll
    for (int ti = 0; ti < 2; ++ti) {
        const int mg = mb * MBLK + wm * 32 + ti * 16 + (hi ? 8 : 0);
        const v4f bv0 = *(const v4f*)&bias[mg];        // mg is 8-aligned
        const v4f bv1 = *(const v4f*)&bias[mg + 4];
#pragma unroll
        for (int tj = 0; tj < 2; ++tj) {
            const int cg = nb * NBLK + wn * 32 + tj * 16 + r16;
            const int n  = cg >> 10;
            const int oh = (cg >> 5) & 31;
            const int ow = cg & 31;
            float* op = out + (((size_t)(n * COUT + mg)) << 10) + (oh << 5) + ow;
#pragma unroll
            for (int r = 0; r < 8; ++r)
                op[(size_t)r << 10] =
                    acc[ti][tj][r] + (r < 4 ? bv0[r] : bv1[r - 4]);
        }
    }
}

extern "C" void kernel_launch(void* const* d_in, const int* in_sizes, int n_in,
                              void* d_out, int out_size, void* d_ws, size_t ws_size,
                              hipStream_t stream) {
    // setup_inputs order: x, time_emb (unused), y (unused), weight, bias
    const float* x    = (const float*)d_in[0];
    const float* w    = (const float*)d_in[3];
    const float* bias = (const float*)d_in[4];
    float*       out  = (float*)d_out;
    _Float16*    wqd  = (_Float16*)d_ws;       // 256*2304*2 B = 1.125 MiB

    quantize_weights<<<dim3(9, COUT, 1), dim3(256, 1, 1), 0, stream>>>(w, wqd);

    dim3 grid(NTOT / NBLK, COUT / MBLK, 1);    // (512, 2)
    ternary_conv_wmma<<<grid, dim3(256, 1, 1), 0, stream>>>(x, wqd, bias, out);
}
